// BodyRayTracing_74577812128125
// MI455X (gfx1250) — compile-verified, matrix operand order checked by reading, not run on hardware
//
#include <hip/hip_runtime.h>

typedef __attribute__((ext_vector_type(16))) _Float16 v16h;
typedef __attribute__((ext_vector_type(8)))  float    v8f;

#define NRAYS    16384
#define O_PTS0   0           // curr_start_points       N*3
#define O_MASK   49152       // network_body_mask       N
#define O_DIS    65536       // acc_start_dis           N
#define O_SPTS   81920       // sampler_pts             N*32*3
#define O_SDIST  1654784     // sampler_dists           N*32
#define O_STRANS 2179072     // sampler_transforms      N*32*16
#define O_SCONV  10567680    // sampler_converge_mask   N

__device__ __forceinline__ float softplusf(float x) {
    // stable: max(x,0) + log1p(exp(-|x|))
    float ax = fabsf(x);
    return fmaxf(x, 0.0f) + __logf(1.0f + __expf(-ax));
}

// ---- T = einsum(mean(skinning_weights)/sum, bone_transforms) -> 16 floats in ws
__global__ void compute_T_kernel(const float* __restrict__ sw,
                                 const float* __restrict__ bt,
                                 float* __restrict__ Tout) {
    __shared__ float sAcc[24];
    int tid = threadIdx.x;
    if (tid < 24) sAcc[tid] = 0.0f;
    __syncthreads();
    float acc[24];
    #pragma unroll
    for (int j = 0; j < 24; ++j) acc[j] = 0.0f;
    for (int v = tid; v < 6890; v += blockDim.x) {
        #pragma unroll
        for (int j = 0; j < 24; ++j) acc[j] += sw[v * 24 + j];
    }
    #pragma unroll
    for (int j = 0; j < 24; ++j) atomicAdd(&sAcc[j], acc[j]);
    __syncthreads();
    if (tid < 16) {
        float tot = 0.0f;
        #pragma unroll
        for (int j = 0; j < 24; ++j) tot += sAcc[j];
        float t = 0.0f;
        #pragma unroll
        for (int j = 0; j < 24; ++j) t += (sAcc[j] / tot) * bt[j * 16 + tid];
        Tout[tid] = t;
    }
}

// ---- broadcast T to (N,32,4,4) with float4 stores: 2,097,152 float4s
__global__ void broadcast_T_kernel(const float* __restrict__ T,
                                   float4* __restrict__ out) {
    __shared__ float4 sT[4];
    if (threadIdx.x < 4) sT[threadIdx.x] = ((const float4*)T)[threadIdx.x];
    __syncthreads();
    size_t i = (size_t)blockIdx.x * blockDim.x + threadIdx.x;
    out[i] = sT[i & 3];
}

// ---- main: 4 waves / block, 16 rays / wave, WMMA MLP
__global__ __launch_bounds__(128) void body_raytrace_kernel(
    const float* __restrict__ g_cam,   const float* __restrict__ g_dirs,
    const float* __restrict__ g_bounds,
    const float* __restrict__ g_loc,   const float* __restrict__ g_sc,
    const float* __restrict__ g_trans, const float* __restrict__ g_cmin,
    const float* __restrict__ g_cmax,  const float* __restrict__ g_center,
    const float* __restrict__ g_W1,    const float* __restrict__ g_b1,
    const float* __restrict__ g_W2,    const float* __restrict__ g_b2,
    const float* __restrict__ g_W3,    const float* __restrict__ g_b3,
    float* __restrict__ out) {
    __shared__ __align__(32) _Float16 sW2T[128 * 128];   // W2 transposed, f16
    __shared__ __align__(32) _Float16 sH[4][16][128];    // per-wave activations
    __shared__ float sX[4][16][3];
    __shared__ float sW1[3 * 128];
    __shared__ float sB1[128], sB2[128], sW3v[128];
    __shared__ float sZ[4][16][32];

    const int tid = threadIdx.x;
    for (int i = tid; i < 128 * 128; i += 128) {
        int n = i >> 7, k = i & 127;
        sW2T[i] = (_Float16)g_W2[(k << 7) + n];
    }
    for (int i = tid; i < 384; i += 128) sW1[i] = g_W1[i];
    if (tid < 128) { sB1[tid] = g_b1[tid]; sB2[tid] = g_b2[tid]; sW3v[tid] = g_W3[tid]; }
    __syncthreads();

    const int w    = tid >> 5;
    const int lane = tid & 31;
    const int r    = lane & 15;
    const int hi   = lane >> 4;
    const int rayBase = (blockIdx.x * 4 + w) * 16;
    const int ray  = rayBase + r;   // lanes l and l+16 duplicate ray state

    const float b3v = g_b3[0];
    float camv[3], tl[3], scv[3], cev[3], ihr[3], dir[3];
    #pragma unroll
    for (int c = 0; c < 3; ++c) {
        camv[c] = g_cam[c];
        tl[c]   = g_trans[c] + g_loc[c];
        scv[c]  = g_sc[c];
        cev[c]  = g_center[c];
        ihr[c]  = 2.0f / (g_cmax[c] - g_cmin[c]);
        dir[c]  = g_dirs[ray * 3 + c];
    }
    const float nearv = g_bounds[ray * 2 + 0];
    const float farv  = g_bounds[ray * 2 + 1];

    auto to_cano = [&](float p, int c) -> float {
        return ((p - tl[c]) * scv[c] - cev[c]) * ihr[c];
    };

    // Full-wave MLP eval at parameter t (uniform trip counts; EXEC all-1 at WMMA)
    auto mlp_eval = [&](float t, float* xn) -> float {
        #pragma unroll
        for (int c = 0; c < 3; ++c) xn[c] = to_cano(camv[c] + t * dir[c], c);
        if (hi == 0) { sX[w][r][0] = xn[0]; sX[w][r][1] = xn[1]; sX[w][r][2] = xn[2]; }
        __syncthreads();
        // layer 1: h1 = softplus(x @ W1 + b1), written f16 row-major
        for (int rr = 0; rr < 16; ++rr) {
            float x0 = sX[w][rr][0], x1 = sX[w][rr][1], x2 = sX[w][rr][2];
            #pragma unroll
            for (int j = 0; j < 4; ++j) {
                int k = lane + 32 * j;
                float h = sB1[k] + x0 * sW1[k] + x1 * sW1[128 + k] + x2 * sW1[256 + k];
                sH[w][rr][k] = (_Float16)softplusf(h);
            }
        }
        __syncthreads();
        // layer 2: 16x128 @ 128x128 via 8 N-tiles x 4 K-steps of WMMA f16
        const _Float16* hrow = &sH[w][r][hi * 16];
        v16h a0 = *(const v16h*)(hrow);
        v16h a1 = *(const v16h*)(hrow + 32);
        v16h a2 = *(const v16h*)(hrow + 64);
        v16h a3 = *(const v16h*)(hrow + 96);
        __syncthreads();   // A in registers; sH may now be overwritten with h2
        #pragma unroll
        for (int nt = 0; nt < 8; ++nt) {
            float bias = sB2[nt * 16 + r];
            v8f c;
            #pragma unroll
            for (int v = 0; v < 8; ++v) c[v] = bias;
            const _Float16* wrow = &sW2T[(nt * 16 + r) * 128 + hi * 16];
            c = __builtin_amdgcn_wmma_f32_16x16x32_f16(false, a0, false, *(const v16h*)(wrow +  0), (short)0, c, false, false);
            c = __builtin_amdgcn_wmma_f32_16x16x32_f16(false, a1, false, *(const v16h*)(wrow + 32), (short)0, c, false, false);
            c = __builtin_amdgcn_wmma_f32_16x16x32_f16(false, a2, false, *(const v16h*)(wrow + 64), (short)0, c, false, false);
            c = __builtin_amdgcn_wmma_f32_16x16x32_f16(false, a3, false, *(const v16h*)(wrow + 96), (short)0, c, false, false);
            #pragma unroll
            for (int v = 0; v < 8; ++v)
                sH[w][v + hi * 8][nt * 16 + r] = (_Float16)softplusf(c[v]);
        }
        __syncthreads();
        // layer 3: s = h2 . W3 + b3 (duplicated in both half-lanes)
        float s = b3v;
        #pragma unroll 8
        for (int k = 0; k < 128; ++k) s += (float)sH[w][r][k] * sW3v[k];
        __syncthreads();
        return s;
    };

    // ---- sphere tracing (32 fixed, masked updates)
    float accd = nearv;
    bool  unf  = nearv < farv;
    bool  dv   = !unf;
    float xn[3];
    #pragma unroll 1
    for (int it = 0; it < 32; ++it) {
        float s  = mlp_eval(accd, xn);
        float sm = fminf(fmaxf(s, -0.1f), 0.1f);
        bool upd = unf && (fabsf(sm) > 1e-5f) && (fabsf(s) < 1e6f);
        float accn = upd ? accd + sm : accd;
        dv = upd ? (accn >= farv) : dv;
        bool remove = (unf && (fabsf(s) <= 1e-5f)) || dv;
        unf = unf && !remove;
        accd = accn;
    }
    // ---- secant refinement (8 fixed)
    float t0 = fminf(fmaxf(accd - 0.1f, nearv), farv);
    float t1 = fminf(fmaxf(accd,        nearv), farv);
    float f0 = mlp_eval(t0, xn);
    #pragma unroll 1
    for (int it = 0; it < 8; ++it) {
        float f1 = mlp_eval(t1, xn);
        float den = f0 - f1;
        den = (fabsf(den) < 1e-9f) ? 1e-9f : den;
        float t2 = t1 - f1 * (t1 - t0) / den;
        t2 = fminf(fmaxf(t2, nearv), farv);
        t0 = t1; f0 = f1; t1 = t2;
    }
    float f1 = mlp_eval(t1, xn);
    bool conv = (fabsf(f1) <= 1e-5f) && (t1 >= nearv) && (t1 <= farv);

    // ---- per-ray outputs + z merge (lanes 0-15 only; no WMMA below)
    if (lane < 16) {
        out[O_PTS0 + ray * 3 + 0] = xn[0];
        out[O_PTS0 + ray * 3 + 1] = xn[1];
        out[O_PTS0 + ray * 3 + 2] = xn[2];
        float cf = conv ? 1.0f : 0.0f;
        out[O_MASK  + ray] = cf;
        out[O_SCONV + ray] = cf;
        out[O_DIS   + ray] = conv ? t1 : nearv;
        // merge two sorted 16-sample ladders (near-surface + stratified)
        float dn  = 0.1f / 15.0f;
        float zn0 = t1 - 0.05f;
        float df  = (farv - nearv) * (1.0f / 15.0f);
        float du  = (farv - nearv) * (1.0f / 31.0f);
        int i = 0, j = 0;
        #pragma unroll 1
        for (int k = 0; k < 32; ++k) {
            float a = (i < 16) ? zn0 + i * dn    : 3.0e38f;
            float b = (j < 16) ? nearv + j * df  : 3.0e38f;
            float zt;
            if (a <= b) { zt = a; ++i; } else { zt = b; ++j; }
            float z = conv ? zt : (nearv + k * du);
            sZ[w][r][k] = z;
            out[O_SDIST + (size_t)ray * 32 + k] = z;
        }
    }
    __syncthreads();
    // ---- sampler_pts: all 32 lanes, lane = sample index, loop over 16 rays
    #pragma unroll 1
    for (int q = 0; q < 16; ++q) {
        int rr = rayBase + q;
        float z = sZ[w][q][lane];
        #pragma unroll
        for (int c = 0; c < 3; ++c) {
            float p = camv[c] + z * g_dirs[rr * 3 + c];
            out[O_SPTS + ((size_t)rr * 32 + lane) * 3 + c] = to_cano(p, c);
        }
    }
}

extern "C" void kernel_launch(void* const* d_in, const int* in_sizes, int n_in,
                              void* d_out, int out_size, void* d_ws, size_t ws_size,
                              hipStream_t stream) {
    (void)in_sizes; (void)n_in; (void)out_size; (void)ws_size;
    const float* g_cam    = (const float*)d_in[0];
    const float* g_dirs   = (const float*)d_in[1];
    const float* g_bounds = (const float*)d_in[2];
    const float* g_loc    = (const float*)d_in[3];
    const float* g_sc     = (const float*)d_in[4];
    const float* g_sw     = (const float*)d_in[7];
    const float* g_bt     = (const float*)d_in[8];
    const float* g_trans  = (const float*)d_in[9];
    const float* g_cmin   = (const float*)d_in[10];
    const float* g_cmax   = (const float*)d_in[11];
    const float* g_center = (const float*)d_in[12];
    const float* g_W1     = (const float*)d_in[13];
    const float* g_b1     = (const float*)d_in[14];
    const float* g_W2     = (const float*)d_in[15];
    const float* g_b2     = (const float*)d_in[16];
    const float* g_W3     = (const float*)d_in[17];
    const float* g_b3     = (const float*)d_in[18];
    float* out = (float*)d_out;
    float* T   = (float*)d_ws;

    compute_T_kernel<<<1, 256, 0, stream>>>(g_sw, g_bt, T);
    body_raytrace_kernel<<<NRAYS / 64, 128, 0, stream>>>(
        g_cam, g_dirs, g_bounds, g_loc, g_sc, g_trans, g_cmin, g_cmax, g_center,
        g_W1, g_b1, g_W2, g_b2, g_W3, g_b3, out);
    // (N*32*16)/4 float4s = 2,097,152 -> 8192 blocks x 256 threads
    broadcast_T_kernel<<<8192, 256, 0, stream>>>(T, (float4*)(out + O_STRANS));
}
